// AttentionOut_33191507263929
// MI455X (gfx1250) — matrix operand order repaired and checked by reading, not compiled, not run on hardware
//
#include <hip/hip_runtime.h>
#include <hip/hip_bf16.h>
#include <math.h>

// ---------------------------------------------------------------------------
// Problem constants (from reference)
// ---------------------------------------------------------------------------
#define N_HEADS 16
#define D_HEAD  64
#define D_MODEL 1024
#define N_CTX   2048
#define NHD     (N_HEADS * D_HEAD)   // 1024, row stride of q/k/v
static_assert(NHD == D_MODEL, "layout assumption");

typedef __attribute__((ext_vector_type(16))) __bf16        v16bf;
typedef __attribute__((ext_vector_type(8)))  float         v8f;
typedef __attribute__((ext_vector_type(4)))  unsigned int  u32x4;

__device__ __forceinline__ __bf16 f2bf(float x) { return (__bf16)x; }

struct frag_bits { u32x4 lo, hi; };
static_assert(sizeof(frag_bits) == sizeof(v16bf), "frag size");

// 16 contiguous bf16 (32B) -> fragment
__device__ __forceinline__ v16bf load_frag16(const __bf16* p) {
    frag_bits fb{*(const u32x4*)p, *(const u32x4*)(p + 8)};
    return __builtin_bit_cast(v16bf, fb);
}
// two separated 8-bf16 runs (16B each) -> fragment
__device__ __forceinline__ v16bf load_frag8x2(const __bf16* p0, const __bf16* p1) {
    frag_bits fb{*(const u32x4*)p0, *(const u32x4*)p1};
    return __builtin_bit_cast(v16bf, fb);
}

__device__ __forceinline__ v8f wmma_bf16(v16bf a, v16bf b, v8f c) {
    // D = A(16x32 bf16) * B(32x16 bf16) + C(16x16 f32)
    return __builtin_amdgcn_wmma_f32_16x16x32_bf16(
        false, a, false, b, (short)0, c, false, false);
}

// ---------------------------------------------------------------------------
// Pre-pass 1: K fp32 -> bf16, same layout (b, kpos, h*64+d)
// ---------------------------------------------------------------------------
__global__ __launch_bounds__(256) void convert_bf16_kernel(
    const float4* __restrict__ in, __bf16* __restrict__ out, int n4)
{
    int i = blockIdx.x * blockDim.x + threadIdx.x;
    const int stride = gridDim.x * blockDim.x;
    for (; i < n4; i += stride) {
        float4 a = in[i];
        out[i * 4 + 0] = f2bf(a.x);
        out[i * 4 + 1] = f2bf(a.y);
        out[i * 4 + 2] = f2bf(a.z);
        out[i * 4 + 3] = f2bf(a.w);
    }
}

// ---------------------------------------------------------------------------
// Pre-pass 2: V (b, k, h*64+d) fp32 -> Vt (b*16+h, d, k) bf16
// ---------------------------------------------------------------------------
__global__ __launch_bounds__(256) void transpose_v_kernel(
    const float* __restrict__ v, __bf16* __restrict__ vt)
{
    __shared__ __bf16 t[64][65];
    const int head = blockIdx.y, b = head >> 4, h = head & 15;
    const int kt = blockIdx.x;

    const float* ip = v + ((size_t)b * N_CTX + (size_t)kt * 64) * NHD + h * D_HEAD;
#pragma unroll
    for (int it = 0; it < 16; ++it) {
        const int idx = it * 256 + threadIdx.x;
        const int r = idx >> 6, c = idx & 63;           // r = key, c = d
        t[r][c] = f2bf(ip[(size_t)r * NHD + c]);
    }
    __syncthreads();
    __bf16* op = vt + (size_t)head * D_HEAD * N_CTX + (size_t)kt * 64;
#pragma unroll
    for (int it = 0; it < 16; ++it) {
        const int idx = it * 256 + threadIdx.x;
        const int c = idx >> 6, r = idx & 63;           // c = d, r = key
        op[(size_t)c * N_CTX + r] = t[r][c];
    }
}

// ---------------------------------------------------------------------------
// Pre-pass 3: W_O (1024 x 1024) fp32 -> Wt (1024 x 1024) bf16 transposed
// ---------------------------------------------------------------------------
__global__ __launch_bounds__(256) void transpose_w_kernel(
    const float* __restrict__ W, __bf16* __restrict__ wt)
{
    __shared__ __bf16 t[64][65];
    const float* ip = W + (size_t)blockIdx.x * 64 * D_MODEL + blockIdx.y * 64;
#pragma unroll
    for (int it = 0; it < 16; ++it) {
        const int idx = it * 256 + threadIdx.x;
        const int r = idx >> 6, c = idx & 63;
        t[r][c] = f2bf(ip[(size_t)r * D_MODEL + c]);
    }
    __syncthreads();
    __bf16* op = wt + (size_t)blockIdx.y * 64 * D_MODEL + blockIdx.x * 64;
#pragma unroll
    for (int it = 0; it < 16; ++it) {
        const int idx = it * 256 + threadIdx.x;
        const int c = idx >> 6, r = idx & 63;
        op[(size_t)c * D_MODEL + r] = t[r][c];
    }
}

// ---------------------------------------------------------------------------
// Kernel: causal flash attention, 64-key blocks, exp2-domain softmax.
//   grid = (N_CTX/64, N_HEADS, B), block = 128 (4 waves).
//   Q is pre-scaled by (1/sqrt(64))*log2(e), so p = exp2(s - m) directly.
//   Blocks strictly below the diagonal run a mask-free fast path.
// ---------------------------------------------------------------------------
__global__ __launch_bounds__(128) void attn_fa_kernel(
    const float* __restrict__ q, const __bf16* __restrict__ kb,
    const __bf16* __restrict__ vt, __bf16* __restrict__ z_out)
{
    __shared__ __attribute__((aligned(16))) __bf16 lds_p[4][16 * 80]; // 16x64, stride 80

    const int wave = threadIdx.x >> 5;
    const int lane = threadIdx.x & 31;
    const int row  = lane & 15;
    const int hi   = lane >> 4;
    const int qtile = blockIdx.x, h = blockIdx.y, b = blockIdx.z;

    const int qbase = qtile * 64 + wave * 16;
    const size_t head_base = ((size_t)b * N_CTX) * NHD + (size_t)h * D_HEAD;
    const __bf16* vhead = vt + ((size_t)(b * N_HEADS + h)) * D_HEAD * N_CTX;

    // ---- Q A-fragments, pre-scaled by (1/sqrt(D)) * log2(e)
    const float qscale = 0.125f * 1.44269504088896f;
    const float* qp = q + head_base + (size_t)(qbase + row) * NHD;
    v16bf aq0, aq1;
#pragma unroll
    for (int i = 0; i < 16; ++i) {
        const int d = i + (hi << 4);
        aq0[i] = f2bf(qp[d] * qscale);
        aq1[i] = f2bf(qp[32 + d] * qscale);
    }

    float m[8], l[8];
    v8f acc[4];
    v8f zf = {};
#pragma unroll
    for (int r = 0; r < 8; ++r) { m[r] = -INFINITY; l[r] = 0.0f; }
#pragma unroll
    for (int j = 0; j < 4; ++j) acc[j] = zf;

    __bf16* pl = &lds_p[wave][0];

    // Shared block body; MASK folds at compile time after inlining.
    auto process_block = [&](int k0, bool MASK) {
        // ---------- S = Q * K^T for 64 keys (4 x 16-key sub-tiles) ----------
        v8f s[4];
#pragma unroll
        for (int c = 0; c < 4; ++c) {
            const __bf16* kp = kb + head_base + (size_t)(k0 + 16 * c + row) * NHD;
            if (!MASK) __builtin_prefetch(kp + (size_t)64 * NHD, 0, 0);
            v16bf bkA = load_frag16(kp + 16 * hi);               // d 0..31
            v16bf bkB = load_frag16(kp + 32 + 16 * hi);          // d 32..63
            s[c] = wmma_bf16(aq0, bkA, zf);
            s[c] = wmma_bf16(aq1, bkB, s[c]);
        }

        // ---------- (mask) + online softmax in exp2 domain ----------
        float mnew[8];
#pragma unroll
        for (int r = 0; r < 8; ++r) {
            const int qpos = qbase + r + (hi << 3);     // C layout M = r + 8*hi
            float t = -INFINITY;
#pragma unroll
            for (int c = 0; c < 4; ++c) {
                float e = s[c][r];
                if (MASK) e = (k0 + 16 * c + row <= qpos) ? e : -INFINITY;
                s[c][r] = e;
                t = fmaxf(t, e);
            }
            t = fmaxf(t, __shfl_xor(t, 1, 32));
            t = fmaxf(t, __shfl_xor(t, 2, 32));
            t = fmaxf(t, __shfl_xor(t, 4, 32));
            t = fmaxf(t, __shfl_xor(t, 8, 32));
            mnew[r] = fmaxf(m[r], t);
        }
#pragma unroll
        for (int r = 0; r < 8; ++r) {
            const float cf = __builtin_exp2f(m[r] - mnew[r]);
            float rs = 0.0f;
            const int M = r + (hi << 3);
#pragma unroll
            for (int c = 0; c < 4; ++c) {
                const float p = __builtin_exp2f(s[c][r] - mnew[r]);
                rs += p;
                pl[M * 80 + 16 * c + row] = f2bf(p);    // C-layout -> LDS
            }
            rs += __shfl_xor(rs, 1, 32);
            rs += __shfl_xor(rs, 2, 32);
            rs += __shfl_xor(rs, 4, 32);
            rs += __shfl_xor(rs, 8, 32);
            l[r] = l[r] * cf + rs;
            m[r] = mnew[r];
#pragma unroll
            for (int j = 0; j < 4; ++j) acc[j][r] *= cf;
        }
        asm volatile("s_wait_dscnt 0" ::: "memory");    // same-wave DS in-order

        // ---------- A-fragments of P ----------
        const __bf16* pr = pl + row * 80;
        v16bf pa0 = load_frag16(pr + (hi << 4));
        v16bf pa1 = load_frag16(pr + 32 + (hi << 4));

        // ---------- Z += P(16x64) * V(64x64) ----------
#pragma unroll
        for (int j = 0; j < 4; ++j) {
            const __bf16* vp = vhead + (size_t)(j * 16 + row) * N_CTX + k0;
            if (!MASK) __builtin_prefetch(vp + 64, 0, 0);
            v16bf bv0 = load_frag16(vp + 16 * hi);          // keys 0..31
            v16bf bv1 = load_frag16(vp + 32 + 16 * hi);     // keys 32..63
            acc[j] = wmma_bf16(pa0, bv0, acc[j]);
            acc[j] = wmma_bf16(pa1, bv1, acc[j]);
        }
    };

    const int kdiag = qtile * 64;            // all waves' rows live in this block
    for (int k0 = 0; k0 < kdiag; k0 += 64)   // strictly-below-diagonal: no mask
        process_block(k0, false);
    process_block(kdiag, true);              // diagonal block: causal mask

    // ---- normalize, store Z bf16 at (b, qpos, h*64+d)
#pragma unroll
    for (int r = 0; r < 8; ++r) {
        const int M = r + (hi << 3);
        const float invl = 1.0f / l[r];
        const size_t o = ((size_t)b * N_CTX + qbase + M) * NHD + (size_t)h * D_HEAD;
#pragma unroll
        for (int j = 0; j < 4; ++j)
            z_out[o + j * 16 + row] = f2bf(acc[j][r] * invl);
    }
}

// ---------------------------------------------------------------------------
// Kernel: out = Z(bf16, BSx1024) @ Wt^T + b_O, fp32 out. No LDS, no barriers.
// ---------------------------------------------------------------------------
__global__ __launch_bounds__(256) void proj_kernel(
    const __bf16* __restrict__ z, const __bf16* __restrict__ wt,
    const float* __restrict__ bO, float* __restrict__ out)
{
    const int wave = threadIdx.x >> 5;
    const int lane = threadIdx.x & 31;
    const int row  = lane & 15;
    const int hi   = lane >> 4;

    const int mrow0 = blockIdx.x * 128 + wave * 16;
    const int ncol0 = blockIdx.y * 64;

    v8f acc[4];
    v8f zf = {};
#pragma unroll
    for (int j = 0; j < 4; ++j) acc[j] = zf;

    const __bf16* zrow = z + (size_t)(mrow0 + row) * D_MODEL;

    for (int kk = 0; kk < D_MODEL; kk += 32) {
        v16bf az = load_frag16(zrow + kk + (hi << 4));
#pragma unroll
        for (int j = 0; j < 4; ++j) {
            const __bf16* wp = wt + (size_t)(ncol0 + j * 16 + row) * D_MODEL
                                  + kk + (hi << 4);
            v16bf bw = load_frag16(wp);
            acc[j] = wmma_bf16(az, bw, acc[j]);
        }
    }

#pragma unroll
    for (int r = 0; r < 8; ++r) {
        const int mrow = mrow0 + r + (hi << 3);
#pragma unroll
        for (int j = 0; j < 4; ++j) {
            const int n = ncol0 + j * 16 + row;
            out[(size_t)mrow * D_MODEL + n] = acc[j][r] + bO[n];
        }
    }
}

// ---------------------------------------------------------------------------
// Residual passthrough (second tuple output)
// ---------------------------------------------------------------------------
__global__ __launch_bounds__(256) void copy_res_kernel(
    const float4* __restrict__ in, float4* __restrict__ out, int n4)
{
    int i = blockIdx.x * blockDim.x + threadIdx.x;
    const int stride = gridDim.x * blockDim.x;
    for (; i < n4; i += stride) out[i] = in[i];
}

// ---------------------------------------------------------------------------
extern "C" void kernel_launch(void* const* d_in, const int* in_sizes, int n_in,
                              void* d_out, int out_size, void* d_ws, size_t ws_size,
                              hipStream_t stream)
{
    const float* q   = (const float*)d_in[0];
    const float* k   = (const float*)d_in[1];
    const float* v   = (const float*)d_in[2];
    const float* res = (const float*)d_in[3];
    const float* W   = (const float*)d_in[4];
    const float* bO  = (const float*)d_in[5];
    float* out = (float*)d_out;

    const int B  = in_sizes[0] / (N_CTX * NHD);   // = 2
    const int BS = B * N_CTX;                     // = 4096

    // workspace carve-up (bytes): Kbf 8MB | Vt 8MB | Wt 2MB | Z 8MB = 26MB
    char* ws = (char*)d_ws;
    __bf16* kbf = (__bf16*)(ws);
    __bf16* vtb = (__bf16*)(ws + (size_t)BS * NHD * 2);
    __bf16* wtb = (__bf16*)(ws + (size_t)BS * NHD * 4);
    __bf16* zb  = (__bf16*)(ws + (size_t)BS * NHD * 4 + (size_t)D_MODEL * D_MODEL * 2);

    convert_bf16_kernel<<<1024, 256, 0, stream>>>(
        (const float4*)k, kbf, (BS * NHD) / 4);
    transpose_v_kernel<<<dim3(N_CTX / 64, B * N_HEADS), 256, 0, stream>>>(v, vtb);
    transpose_w_kernel<<<dim3(16, 16), 256, 0, stream>>>(W, wtb);

    attn_fa_kernel<<<dim3(N_CTX / 64, N_HEADS, B), 128, 0, stream>>>(q, kbf, vtb, zb);

    proj_kernel<<<dim3(BS / 128, D_MODEL / 64), 256, 0, stream>>>(zb, wtb, bO, out);

    copy_res_kernel<<<1024, 256, 0, stream>>>(
        (const float4*)res, (float4*)(out + (size_t)BS * D_MODEL), (BS * D_MODEL) / 4);
}
